// Mamba2Mixer_36532991820117
// MI455X (gfx1250) — compile-verified, matrix operand order checked by reading, not run on hardware
//
#include <hip/hip_runtime.h>
#include <hip/hip_bf16.h>

// ---------------- problem constants ----------------
constexpr int L        = 4096;
constexpr int HID      = 2048;
constexpr int H        = 64;
constexpr int P        = 64;
constexpr int N        = 128;
constexpr int CH       = 256;
constexpr int KCONV    = 4;
constexpr int INTER    = H * P;             // 4096
constexpr int CONV_DIM = INTER + 2 * N;     // 4352
constexpr int OUT_IN   = 2 * (INTER + N) + H; // 8512
constexpr int NC       = L / CH;            // 16
constexpr float EPSF   = 1e-5f;

typedef __attribute__((ext_vector_type(16))) __bf16 v16bf;
typedef __attribute__((ext_vector_type(8)))  float  v8f;
typedef __attribute__((ext_vector_type(2)))  float  v2f;

#define DEV static __device__ __forceinline__

DEV v8f vzero8() {
  v8f z;
#pragma unroll
  for (int i = 0; i < 8; ++i) z[i] = 0.f;
  return z;
}

DEV v8f wmma_bf16(v16bf a, v16bf b, v8f c) {
  return __builtin_amdgcn_wmma_f32_16x16x32_bf16(false, a, false, b, (short)0, c,
                                                 false, false);
}
DEV v8f wmma_f32k4(v2f a, v2f b, v8f c) {
  return __builtin_amdgcn_wmma_f32_16x16x4_f32(false, a, false, b, (short)0, c,
                                               false, false);
}

DEV float siluf(float x) { return x / (1.f + __expf(-x)); }

// =====================================================================
// Generic bf16 WMMA GEMM: C[M][Nn] = A[M][Kd] * B[Kd][Nn], fp32 in/out.
// BM=128, BN=64, BK=32, 256 threads = 8 waves (2x4 wave grid, 32x32 each).
// =====================================================================
__global__ void gemm_bf16_wmma(const float* __restrict__ A, const float* __restrict__ B,
                               float* __restrict__ C, int M, int Nn, int Kd,
                               int lda, int ldb, int ldc)
{
  __shared__ __bf16 lA[128][34];
  __shared__ __bf16 lB[64][34];
  const int tid  = threadIdx.x;
  const int wave = tid >> 5;
  const int lane = tid & 31;
  const int r16  = lane & 15;
  const int half = lane >> 4;
  const int wy = wave >> 1;   // 0..3 : M sub-block of 32
  const int wx = wave & 1;    // 0..1 : N sub-block of 32
  const int m0 = blockIdx.y * 128;
  const int n0 = blockIdx.x * 64;

  v8f acc[2][2];
#pragma unroll
  for (int i = 0; i < 2; ++i)
#pragma unroll
    for (int j = 0; j < 2; ++j) acc[i][j] = vzero8();

  for (int kk = 0; kk < Kd; kk += 32) {
    // stage A 128x32 (fp32 -> bf16)
    for (int i = tid; i < 128 * 32; i += 256) {
      int r = i >> 5, c = i & 31;
      lA[r][c] = (__bf16)A[(size_t)(m0 + r) * lda + (kk + c)];
    }
    // stage B transposed: lB[n][k]
    for (int i = tid; i < 64 * 32; i += 256) {
      int k = i >> 6, n = i & 63;
      lB[n][k] = (__bf16)B[(size_t)(kk + k) * ldb + (n0 + n)];
    }
    __syncthreads();

    v16bf afr[2], bfr[2];
#pragma unroll
    for (int t = 0; t < 2; ++t) {
      const int mrow = wy * 32 + t * 16 + r16;
      const int ncol = wx * 32 + t * 16 + r16;
#pragma unroll
      for (int v = 0; v < 8; ++v) {
        const int kb = (v < 4) ? (half * 8 + 2 * v) : (16 + half * 8 + 2 * (v - 4));
        afr[t][2 * v]     = lA[mrow][kb];
        afr[t][2 * v + 1] = lA[mrow][kb + 1];
        bfr[t][2 * v]     = lB[ncol][kb];
        bfr[t][2 * v + 1] = lB[ncol][kb + 1];
      }
    }
#pragma unroll
    for (int i = 0; i < 2; ++i)
#pragma unroll
      for (int j = 0; j < 2; ++j)
        acc[i][j] = wmma_bf16(afr[i], bfr[j], acc[i][j]);
    __syncthreads();
  }

#pragma unroll
  for (int i = 0; i < 2; ++i)
#pragma unroll
    for (int j = 0; j < 2; ++j)
#pragma unroll
      for (int v = 0; v < 8; ++v) {
        const int row = m0 + wy * 32 + i * 16 + half * 8 + v;
        const int col = n0 + wx * 32 + j * 16 + r16;
        C[(size_t)row * ldc + col] = acc[i][j][v];
      }
}

// =====================================================================
// Generic batched fp32 WMMA GEMM (used for G = C_chunk @ B_chunk^T).
// transB: B operand element (k,n) = B[n*ldb + k].
// =====================================================================
__global__ void gemm_f32_wmma(const float* __restrict__ Abase, const float* __restrict__ Bbase,
                              float* __restrict__ Cbase, int M, int Nn, int Kd,
                              int lda, int ldb, int ldc, int transB,
                              size_t strideA, size_t strideB, size_t strideC)
{
  const float* A = Abase + (size_t)blockIdx.z * strideA;
  const float* B = Bbase + (size_t)blockIdx.z * strideB;
  float* C       = Cbase + (size_t)blockIdx.z * strideC;

  __shared__ float lA[128][33];
  __shared__ float lB[64][33];
  const int tid  = threadIdx.x;
  const int wave = tid >> 5;
  const int lane = tid & 31;
  const int r16  = lane & 15;
  const int half = lane >> 4;
  const int wy = wave >> 1;
  const int wx = wave & 1;
  const int m0 = blockIdx.y * 128;
  const int n0 = blockIdx.x * 64;

  v8f acc[2][2];
#pragma unroll
  for (int i = 0; i < 2; ++i)
#pragma unroll
    for (int j = 0; j < 2; ++j) acc[i][j] = vzero8();

  for (int kk = 0; kk < Kd; kk += 32) {
    for (int i = tid; i < 128 * 32; i += 256) {
      int r = i >> 5, c = i & 31;
      lA[r][c] = A[(size_t)(m0 + r) * lda + (kk + c)];
    }
    for (int i = tid; i < 64 * 32; i += 256) {
      int k = i >> 6, n = i & 63;
      lB[n][k] = transB ? B[(size_t)(n0 + n) * ldb + (kk + k)]
                        : B[(size_t)(kk + k) * ldb + (n0 + n)];
    }
    __syncthreads();
#pragma unroll
    for (int ks = 0; ks < 32; ks += 4) {
      v2f af[2], bf[2];
#pragma unroll
      for (int t = 0; t < 2; ++t) {
#pragma unroll
        for (int v = 0; v < 2; ++v) {
          af[t][v] = lA[wy * 32 + t * 16 + r16][ks + half * 2 + v];
          bf[t][v] = lB[wx * 32 + t * 16 + r16][ks + half * 2 + v];
        }
      }
#pragma unroll
      for (int i = 0; i < 2; ++i)
#pragma unroll
        for (int j = 0; j < 2; ++j)
          acc[i][j] = wmma_f32k4(af[i], bf[j], acc[i][j]);
    }
    __syncthreads();
  }

#pragma unroll
  for (int i = 0; i < 2; ++i)
#pragma unroll
    for (int j = 0; j < 2; ++j)
#pragma unroll
      for (int v = 0; v < 8; ++v) {
        const int row = m0 + wy * 32 + i * 16 + half * 8 + v;
        const int col = n0 + wx * 32 + j * 16 + r16;
        C[(size_t)row * ldc + col] = acc[i][j][v];
      }
}

// =====================================================================
// Causal depthwise conv (K=4) + SiLU on xBC columns of zxbcdt.
// =====================================================================
__global__ void conv_silu_kernel(const float* __restrict__ zx,
                                 const float* __restrict__ conv_w,
                                 const float* __restrict__ conv_b,
                                 float* __restrict__ xbc)
{
  const long long idx = (long long)blockIdx.x * blockDim.x + threadIdx.x;
  if (idx >= (long long)L * CONV_DIM) return;
  const int l = (int)(idx / CONV_DIM);
  const int c = (int)(idx % CONV_DIM);
  float acc = conv_b[c];
#pragma unroll
  for (int k = 0; k < KCONV; ++k) {
    const int ls = l - (KCONV - 1) + k;
    if (ls >= 0)
      acc += zx[(size_t)ls * OUT_IN + INTER + c] * conv_w[k * CONV_DIM + c];
  }
  xbc[(size_t)l * CONV_DIM + c] = siluf(acc);
}

// dt_proc[l][h] = clip(softplus(dt + dt_bias), 0, 100)
__global__ void dt_kernel(const float* __restrict__ zx, const float* __restrict__ dt_bias,
                          float* __restrict__ dtp)
{
  const int idx = blockIdx.x * blockDim.x + threadIdx.x;
  if (idx >= L * H) return;
  const int l = idx / H, h = idx % H;
  const float v = zx[(size_t)l * OUT_IN + (2 * INTER + 2 * N) + h] + dt_bias[h];
  float sp = (v > 20.f) ? v : log1pf(__expf(v));
  sp = fminf(fmaxf(sp, 0.f), 100.f);
  dtp[(size_t)l * H + h] = sp;
}

// A_cs[h][c][l] = cumsum_l (dt*A), chunk_end[h][c] = last value
__global__ void cumsum_kernel(const float* __restrict__ dtp, const float* __restrict__ A_log,
                              float* __restrict__ acs, float* __restrict__ cend)
{
  const int idx = blockIdx.x * blockDim.x + threadIdx.x;
  if (idx >= H * NC) return;
  const int h = idx / NC, c = idx % NC;
  const float A = -__expf(A_log[h]);
  float cs = 0.f;
  for (int l = 0; l < CH; ++l) {
    cs += dtp[(size_t)(c * CH + l) * H + h] * A;
    acs[((size_t)h * NC + c) * CH + l] = cs;
  }
  cend[h * NC + c] = cs;
}

// =====================================================================
// Y_diag per (head, chunk): y = (G .* exp(acs_l-acs_s) .* tril) @ (x*dt) + x*D
// grid (H, NC), 256 threads. Wave w owns l-rows [32w,32w+32), all 64 p.
// =====================================================================
__global__ void ydiag_kernel(const float* __restrict__ G, const float* __restrict__ xbc,
                             const float* __restrict__ dtp, const float* __restrict__ acs,
                             const float* __restrict__ Dv, float* __restrict__ y)
{
  const int h = blockIdx.x, c = blockIdx.y;
  __shared__ float acs_s[CH];
  __shared__ float dt_s[CH];
  const int tid = threadIdx.x;
  acs_s[tid] = acs[((size_t)h * NC + c) * CH + tid];
  dt_s[tid]  = dtp[(size_t)(c * CH + tid) * H + h];
  __syncthreads();

  const int wave = tid >> 5, lane = tid & 31, r16 = lane & 15, half = lane >> 4;
  const float* Gc = G + (size_t)c * CH * CH;
  const float* xc = xbc + (size_t)c * CH * CONV_DIM + h * P;

  v8f acc[2][4];
#pragma unroll
  for (int i = 0; i < 2; ++i)
#pragma unroll
    for (int j = 0; j < 4; ++j) acc[i][j] = vzero8();

  for (int s0 = 0; s0 < CH; s0 += 4) {
    const int sr = s0 + half * 2;
    v2f bf[4];
#pragma unroll
    for (int j = 0; j < 4; ++j) {
      const int p = j * 16 + r16;
      bf[j][0] = xc[(size_t)sr * CONV_DIM + p] * dt_s[sr];
      bf[j][1] = xc[(size_t)(sr + 1) * CONV_DIM + p] * dt_s[sr + 1];
    }
    v2f af[2];
#pragma unroll
    for (int i = 0; i < 2; ++i) {
      const int l = wave * 32 + i * 16 + r16;
      const float al = acs_s[l];
#pragma unroll
      for (int v = 0; v < 2; ++v) {
        const int s = sr + v;
        af[i][v] = (s <= l) ? Gc[(size_t)l * CH + s] * __expf(al - acs_s[s]) : 0.f;
      }
    }
#pragma unroll
    for (int i = 0; i < 2; ++i)
#pragma unroll
      for (int j = 0; j < 4; ++j)
        acc[i][j] = wmma_f32k4(af[i], bf[j], acc[i][j]);
  }

  const float Dh = Dv[h];
#pragma unroll
  for (int i = 0; i < 2; ++i)
#pragma unroll
    for (int j = 0; j < 4; ++j)
#pragma unroll
      for (int v = 0; v < 8; ++v) {
        const int l = wave * 32 + i * 16 + half * 8 + v;
        const int p = j * 16 + r16;
        const float xv = xc[(size_t)l * CONV_DIM + p];
        y[(size_t)(c * CH + l) * INTER + h * P + p] = acc[i][j][v] + xv * Dh;
      }
}

// =====================================================================
// states per (head, chunk): states[p][n] = sum_l x[l][p]*dt[l]*decay[l]*B[l][n]
// grid (H, NC). Wave: p-tile = w&3, n half = w>>2 (64 cols of N).
// =====================================================================
__global__ void states_kernel(const float* __restrict__ xbc, const float* __restrict__ dtp,
                              const float* __restrict__ acs, float* __restrict__ st)
{
  const int h = blockIdx.x, c = blockIdx.y;
  __shared__ float wdec[CH]; // dt[l] * exp(acs_end - acs[l])
  const int tid = threadIdx.x;
  {
    const float al   = acs[((size_t)h * NC + c) * CH + tid];
    const float aend = acs[((size_t)h * NC + c) * CH + CH - 1];
    wdec[tid] = __expf(aend - al) * dtp[(size_t)(c * CH + tid) * H + h];
  }
  __syncthreads();

  const int wave = tid >> 5, lane = tid & 31, r16 = lane & 15, half = lane >> 4;
  const int pt = wave & 3;
  const int nb = (wave >> 2) * 64;
  const float* xc = xbc + (size_t)c * CH * CONV_DIM + h * P;
  const float* Bc = xbc + (size_t)c * CH * CONV_DIM + INTER;

  v8f acc[4];
#pragma unroll
  for (int j = 0; j < 4; ++j) acc[j] = vzero8();

  for (int l0 = 0; l0 < CH; l0 += 4) {
    const int lr = l0 + half * 2;
    v2f af;
    af[0] = xc[(size_t)lr * CONV_DIM + pt * 16 + r16] * wdec[lr];
    af[1] = xc[(size_t)(lr + 1) * CONV_DIM + pt * 16 + r16] * wdec[lr + 1];
    v2f bf[4];
#pragma unroll
    for (int j = 0; j < 4; ++j) {
      const int n = nb + j * 16 + r16;
      bf[j][0] = Bc[(size_t)lr * CONV_DIM + n];
      bf[j][1] = Bc[(size_t)(lr + 1) * CONV_DIM + n];
    }
#pragma unroll
    for (int j = 0; j < 4; ++j) acc[j] = wmma_f32k4(af, bf[j], acc[j]);
  }

  float* sc = st + ((size_t)c * H + h) * P * N;
#pragma unroll
  for (int j = 0; j < 4; ++j)
#pragma unroll
    for (int v = 0; v < 8; ++v) {
      const int p = pt * 16 + half * 8 + v;
      const int n = nb + j * 16 + r16;
      sc[(size_t)p * N + n] = acc[j][v];
    }
}

// inter-chunk recurrence: states_in[c] = carry; carry = carry*exp(cend[c]) + states[c]
__global__ void chunkscan_kernel(const float* __restrict__ st, const float* __restrict__ cend,
                                 float* __restrict__ stin)
{
  const int idx = blockIdx.x * blockDim.x + threadIdx.x;
  if (idx >= H * P * N) return;
  const int h = idx / (P * N);
  float carry = 0.f;
  for (int c = 0; c < NC; ++c) {
    const size_t s = (size_t)c * H * P * N + idx;
    stin[s] = carry;
    carry = carry * __expf(cend[h * NC + c]) + st[s];
  }
}

// =====================================================================
// Y_off per (head, chunk): y += exp(acs[l]) * C[l,:] @ states_in[p,:]^T
// =====================================================================
__global__ void yoff_kernel(const float* __restrict__ xbc, const float* __restrict__ stin,
                            const float* __restrict__ acs, float* __restrict__ y)
{
  const int h = blockIdx.x, c = blockIdx.y;
  __shared__ float acs_s[CH];
  const int tid = threadIdx.x;
  acs_s[tid] = acs[((size_t)h * NC + c) * CH + tid];
  __syncthreads();

  const int wave = tid >> 5, lane = tid & 31, r16 = lane & 15, half = lane >> 4;
  const float* Cc = xbc + (size_t)c * CH * CONV_DIM + INTER + N;
  const float* S  = stin + ((size_t)c * H + h) * P * N;

  v8f acc[2][4];
#pragma unroll
  for (int i = 0; i < 2; ++i)
#pragma unroll
    for (int j = 0; j < 4; ++j) acc[i][j] = vzero8();

  for (int n0 = 0; n0 < N; n0 += 4) {
    const int nr = n0 + half * 2;
    v2f af[2];
#pragma unroll
    for (int i = 0; i < 2; ++i) {
      const int l = wave * 32 + i * 16 + r16;
      af[i][0] = Cc[(size_t)l * CONV_DIM + nr];
      af[i][1] = Cc[(size_t)l * CONV_DIM + nr + 1];
    }
    v2f bf[4];
#pragma unroll
    for (int j = 0; j < 4; ++j) {
      const int p = j * 16 + r16;
      bf[j][0] = S[(size_t)p * N + nr];
      bf[j][1] = S[(size_t)p * N + nr + 1];
    }
#pragma unroll
    for (int i = 0; i < 2; ++i)
#pragma unroll
      for (int j = 0; j < 4; ++j)
        acc[i][j] = wmma_f32k4(af[i], bf[j], acc[i][j]);
  }

#pragma unroll
  for (int i = 0; i < 2; ++i)
#pragma unroll
    for (int j = 0; j < 4; ++j)
#pragma unroll
      for (int v = 0; v < 8; ++v) {
        const int l = wave * 32 + i * 16 + half * 8 + v;
        const int p = j * 16 + r16;
        const size_t o = (size_t)(c * CH + l) * INTER + h * P + p;
        y[o] += __expf(acs_s[l]) * acc[i][j][v];
      }
}

// gated RMSNorm: yf = y*silu(z); yf *= rsqrt(mean(yf^2)+eps)*norm_w
__global__ void gate_norm_kernel(const float* __restrict__ y, const float* __restrict__ zx,
                                 const float* __restrict__ nw, float* __restrict__ yn)
{
  const int l = blockIdx.x;
  __shared__ float red[256];
  const float* yr = y + (size_t)l * INTER;
  const float* zr = zx + (size_t)l * OUT_IN; // z = first INTER cols
  float vals[16];
  float ss = 0.f;
#pragma unroll
  for (int i = 0; i < 16; ++i) {
    const int c = threadIdx.x + i * 256;
    const float v = yr[c] * siluf(zr[c]);
    vals[i] = v;
    ss += v * v;
  }
  red[threadIdx.x] = ss;
  __syncthreads();
  for (int s = 128; s > 0; s >>= 1) {
    if (threadIdx.x < s) red[threadIdx.x] += red[threadIdx.x + s];
    __syncthreads();
  }
  const float scale = rsqrtf(red[0] / (float)INTER + EPSF);
#pragma unroll
  for (int i = 0; i < 16; ++i) {
    const int c = threadIdx.x + i * 256;
    yn[(size_t)l * INTER + c] = vals[i] * scale * nw[c];
  }
}

// =====================================================================
extern "C" void kernel_launch(void* const* d_in, const int* in_sizes, int n_in,
                              void* d_out, int out_size, void* d_ws, size_t ws_size,
                              hipStream_t stream)
{
  (void)in_sizes; (void)n_in; (void)out_size; (void)ws_size;
  const float* hidden  = (const float*)d_in[0];
  const float* W_in    = (const float*)d_in[1];
  const float* conv_w  = (const float*)d_in[2];
  const float* conv_b  = (const float*)d_in[3];
  const float* dt_bias = (const float*)d_in[4];
  const float* A_log   = (const float*)d_in[5];
  const float* Dv      = (const float*)d_in[6];
  const float* norm_w  = (const float*)d_in[7];
  const float* W_out   = (const float*)d_in[8];
  float* out = (float*)d_out;

  float* ws = (float*)d_ws;
  size_t o = 0;
  float* zx    = ws + o; o += (size_t)L * OUT_IN;     // 34.9M
  float* xbc   = ws + o; o += (size_t)L * CONV_DIM;   // 17.8M
  float* dtp   = ws + o; o += (size_t)L * H;
  float* acs   = ws + o; o += (size_t)L * H;
  float* cend  = ws + o; o += (size_t)H * NC;
  float* G     = ws + o; o += (size_t)NC * CH * CH;
  float* st    = ws + o; o += (size_t)NC * H * P * N;
  float* stin  = ws + o; o += (size_t)NC * H * P * N;
  float* ybuf  = ws + o; o += (size_t)L * INTER;
  float* ynorm = ws + o; o += (size_t)L * INTER;

  // 1. zxbcdt = hidden @ W_in  (bf16 WMMA, fp32 accum)
  gemm_bf16_wmma<<<dim3(OUT_IN / 64, L / 128), 256, 0, stream>>>(
      hidden, W_in, zx, L, OUT_IN, HID, HID, OUT_IN, OUT_IN);
  // 2. conv + silu
  conv_silu_kernel<<<(int)(((size_t)L * CONV_DIM + 255) / 256), 256, 0, stream>>>(
      zx, conv_w, conv_b, xbc);
  // 3. dt processing
  dt_kernel<<<(L * H + 255) / 256, 256, 0, stream>>>(zx, dt_bias, dtp);
  // 4. per-(h,chunk) cumsum
  cumsum_kernel<<<(H * NC + 255) / 256, 256, 0, stream>>>(dtp, A_log, acs, cend);
  // 5. G_c = C_c @ B_c^T  (batched fp32 WMMA)
  gemm_f32_wmma<<<dim3(CH / 64, CH / 128, NC), 256, 0, stream>>>(
      xbc + INTER + N, xbc + INTER, G, CH, CH, N, CONV_DIM, CONV_DIM, CH, 1,
      (size_t)CH * CONV_DIM, (size_t)CH * CONV_DIM, (size_t)CH * CH);
  // 6. Y_diag (+ x*D) -> ybuf
  ydiag_kernel<<<dim3(H, NC), 256, 0, stream>>>(G, xbc, dtp, acs, Dv, ybuf);
  // 7. per-chunk states
  states_kernel<<<dim3(H, NC), 256, 0, stream>>>(xbc, dtp, acs, st);
  // 8. inter-chunk recurrence
  chunkscan_kernel<<<(H * P * N + 255) / 256, 256, 0, stream>>>(st, cend, stin);
  // 9. Y_off accumulate into ybuf
  yoff_kernel<<<dim3(H, NC), 256, 0, stream>>>(xbc, stin, acs, ybuf);
  // 10. gated RMSNorm
  gate_norm_kernel<<<L, 256, 0, stream>>>(ybuf, zx, norm_w, ynorm);
  // 11. out = ynorm @ W_out (bf16 WMMA)
  gemm_bf16_wmma<<<dim3(HID / 64, L / 128), 256, 0, stream>>>(
      ynorm, W_out, out, L, HID, INTER, INTER, HID, HID);
}
